// ViTRPECell_15101105013529
// MI455X (gfx1250) — compile-verified, matrix operand order checked by reading, not run on hardware
//
#include <hip/hip_runtime.h>
#include <hip/hip_bf16.h>
#include <cstdint>
#include <cstddef>

typedef __attribute__((ext_vector_type(16))) _Float16 v16h;
typedef __attribute__((ext_vector_type(8)))  _Float16 v8h;
typedef __attribute__((ext_vector_type(8)))  float    v8f;
typedef __attribute__((ext_vector_type(4)))  int      v4i;
typedef _Float16 half_t;

#define NB    8
#define CDIM  384
#define LTOK  1024
#define HEADS 8
#define HD    48
#define HDP   64
#define DMLP  1536
#define MTOT  (NB*LTOK)           // 8192 rows (tokens)
#define QKVSZ ((size_t)NB*HEADS*LTOK*HDP)  // elements per q/k/v tensor

// ---------------------------------------------------------------------------
// Async global->LDS copy (gfx1250 GLOBAL_LOAD_ASYNC_TO_LDS_B128, ASYNCcnt).
// Guarded: falls back to plain LDS stores if the builtin is not declared.
// The builtin's params are int4 pointers (AS1 global, AS3 LDS).
// ---------------------------------------------------------------------------
#if defined(__has_builtin)
#if __has_builtin(__builtin_amdgcn_global_load_async_to_lds_b128) && \
    __has_builtin(__builtin_amdgcn_s_wait_asynccnt)
#define USE_ASYNC_LDS 1
#else
#define USE_ASYNC_LDS 0
#endif
#else
#define USE_ASYNC_LDS 0
#endif

typedef __attribute__((address_space(1))) v4i gv4i_t;
typedef __attribute__((address_space(3))) v4i lv4i_t;

__device__ __forceinline__ void async_cp16(void* lds_dst, const void* gsrc) {
#if USE_ASYNC_LDS
  __builtin_amdgcn_global_load_async_to_lds_b128(
      (gv4i_t*)(uintptr_t)gsrc, (lv4i_t*)(uint32_t)(uintptr_t)lds_dst, 0, 0);
#else
  *(v8h*)lds_dst = *(const v8h*)gsrc;
#endif
}
__device__ __forceinline__ void async_wait() {
#if USE_ASYNC_LDS
  __builtin_amdgcn_s_wait_asynccnt(0);
#endif
}

// ---------------------------------------------------------------------------
// WMMA helpers (gfx1250, wave32, 16x16x32 f16 -> f32)
// ---------------------------------------------------------------------------
__device__ __forceinline__ v8f wmma_f16(v16h a, v16h b, v8f c) {
  return __builtin_amdgcn_wmma_f32_16x16x32_f16(false, a, false, b, (short)0, c,
                                                false, false);
}

// One 16x32 A/B fragment for this lane (CDNA5 layout): two 16B chunks at
// half-offsets kb and kb+16, kb = (lane>>4)*8.
__device__ __forceinline__ v16h frag_ld(const half_t* p, int lane) {
  const int kb = (lane >> 4) << 3;
  v8h a = *(const v8h*)(p + kb);
  v8h b = *(const v8h*)(p + kb + 16);
  v16h r;
#pragma unroll
  for (int i = 0; i < 8; ++i) { r[i] = a[i]; r[i + 8] = b[i]; }
  return r;
}

__device__ __forceinline__ v8f v8f_zero() {
  v8f z = {0.f, 0.f, 0.f, 0.f, 0.f, 0.f, 0.f, 0.f};
  return z;
}

// ---------------------------------------------------------------------------
// Utility kernels
// ---------------------------------------------------------------------------
__global__ void fill0_kernel(uint32_t* __restrict__ p, size_t n32) {
  size_t i = (size_t)blockIdx.x * 256 + threadIdx.x;
  if (i < n32) p[i] = 0u;
}

// in: float [K][Nc] row-major  ->  out: half [Nc][K] row-major (K-contiguous)
__global__ void tconv_kernel(const float* __restrict__ in, half_t* __restrict__ out,
                             int K, int Nc) {
  int i = blockIdx.x * 256 + threadIdx.x;
  if (i >= K * Nc) return;
  int n = i / K;
  int k = i - n * K;
  out[i] = (half_t)in[(size_t)k * Nc + n];
}

// LayerNorm over C of NCHW x, add pos_embed, emit f16 token-major [8192][384].
__global__ __launch_bounds__(256) void ln1_kernel(const float* __restrict__ x,
                                                  const float* __restrict__ g,
                                                  const float* __restrict__ b,
                                                  const float* __restrict__ pos,
                                                  half_t* __restrict__ xf) {
  int t = blockIdx.x * 256 + threadIdx.x;          // 0..8191
  int n = t >> 10, l = t & 1023;
  const float* base = x + (size_t)n * CDIM * LTOK + l;
  float s = 0.f, ss = 0.f;
  for (int c = 0; c < CDIM; ++c) {
    float v = base[(size_t)c * LTOK];
    s += v; ss += v * v;
  }
  float mean = s * (1.f / CDIM);
  float var  = ss * (1.f / CDIM) - mean * mean;
  float rstd = rsqrtf(var + 1e-5f);
  for (int c = 0; c < CDIM; ++c) {
    float v = base[(size_t)c * LTOK];
    float y = (v - mean) * rstd * g[c] + b[c] + pos[(size_t)l * CDIM + c];
    xf[(size_t)t * CDIM + c] = (half_t)y;
  }
}

// LayerNorm over last axis of token-major x2 -> f16 y2 (wave per token).
__global__ __launch_bounds__(256) void ln2_kernel(const float* __restrict__ x2,
                                                  const float* __restrict__ g,
                                                  const float* __restrict__ b,
                                                  half_t* __restrict__ y2) {
  int lane = threadIdx.x & 31, wv = threadIdx.x >> 5;
  int t = blockIdx.x * 8 + wv;
  const float* row = x2 + (size_t)t * CDIM;
  float s = 0.f, ss = 0.f;
  for (int c = lane; c < CDIM; c += 32) {
    float v = row[c];
    s += v; ss += v * v;
  }
#pragma unroll
  for (int m = 1; m < 32; m <<= 1) {
    s  += __shfl_xor(s,  m, 32);
    ss += __shfl_xor(ss, m, 32);
  }
  float mean = s * (1.f / CDIM);
  float var  = ss * (1.f / CDIM) - mean * mean;
  float rstd = rsqrtf(var + 1e-5f);
  for (int c = lane; c < CDIM; c += 32)
    y2[(size_t)t * CDIM + c] = (half_t)((row[c] - mean) * rstd * g[c] + b[c]);
}

// ---------------------------------------------------------------------------
// Generic WMMA GEMM: C[M,Nc] = A[M,K](f16) * Bt[Nc,K](f16)^T + bias, fused
// epilogues.  Block tile 128x128, BK=32, 8 waves (wave tile 64x32 = 8 WMMA).
// Tiles staged with async global->LDS copies where available.
// ---------------------------------------------------------------------------
#define EPI_QKV  0   // scatter into padded q/k/v (half), +b_qkv
#define EPI_PROJ 1   // x2 = x(NCHW) + (.+b_proj), float token-major
#define EPI_GELU 2   // half = gelu(.+b_mlp1), row-major [M][Nc]
#define EPI_MLP2 3   // d_out(NCHW) = x2 + (.+b_mlp2)

template <int EPI>
__global__ __launch_bounds__(256) void gemm_wmma(const half_t* __restrict__ A,
                                                 const half_t* __restrict__ Bt,
                                                 const float* __restrict__ bias,
                                                 int M, int Nc, int K,
                                                 const float* __restrict__ res,
                                                 half_t* __restrict__ out_h,
                                                 float* __restrict__ out_f) {
  __shared__ __align__(16) half_t Alds[128 * 32];
  __shared__ __align__(16) half_t Blds[128 * 32];

  const int tid  = threadIdx.x;
  const int lane = tid & 31;
  const int wv   = tid >> 5;
  const int wm   = (wv >> 2) * 64;   // wave row offset in tile
  const int wn   = (wv & 3) * 32;    // wave col offset in tile
  const int bm   = blockIdx.x * 128;
  const int bn   = blockIdx.y * 128;
  const int mrow = lane & 15;

  v8f acc[4][2];
#pragma unroll
  for (int i = 0; i < 4; ++i)
#pragma unroll
    for (int j = 0; j < 2; ++j) acc[i][j] = v8f_zero();

  const int steps = K >> 5;
  for (int k0 = 0; k0 < steps; ++k0) {
#pragma unroll
    for (int i = 0; i < 2; ++i) {
      int ch = tid + i * 256;        // 512 x 16B chunks per tile
      int r  = ch >> 2;
      int c  = (ch & 3) * 8;
      async_cp16(Alds + r * 32 + c, A + (size_t)(bm + r) * K + k0 * 32 + c);
      async_cp16(Blds + r * 32 + c, Bt + (size_t)(bn + r) * K + k0 * 32 + c);
    }
    if (k0 + 1 < steps) {            // global_prefetch_b8 hints for next tiles
      __builtin_prefetch(A + (size_t)(bm + (tid >> 1)) * K + (k0 + 1) * 32, 0, 1);
      __builtin_prefetch(Bt + (size_t)(bn + (tid >> 1)) * K + (k0 + 1) * 32, 0, 1);
    }
    async_wait();
    __syncthreads();

    v16h bfr[2];
#pragma unroll
    for (int nt = 0; nt < 2; ++nt)
      bfr[nt] = frag_ld(Blds + (wn + nt * 16 + mrow) * 32, lane);
#pragma unroll
    for (int mt = 0; mt < 4; ++mt) {
      v16h af = frag_ld(Alds + (wm + mt * 16 + mrow) * 32, lane);
      acc[mt][0] = wmma_f16(af, bfr[0], acc[mt][0]);
      acc[mt][1] = wmma_f16(af, bfr[1], acc[mt][1]);
    }
    __syncthreads();
  }

  // Epilogue: C layout — lane holds N = lane&15, rows M = (lane>>4)*8 + j.
  const int rbase = bm + wm + ((lane >> 4) << 3);
#pragma unroll
  for (int mt = 0; mt < 4; ++mt)
#pragma unroll
    for (int nt = 0; nt < 2; ++nt)
#pragma unroll
      for (int j = 0; j < 8; ++j) {
        int row = rbase + mt * 16 + j;
        int col = bn + wn + nt * 16 + (lane & 15);
        float v = acc[mt][nt][j] + bias[col];
        if (EPI == EPI_QKV) {
          int m3  = col / CDIM;
          int rem = col - m3 * CDIM;
          int h   = rem / HD;
          int dd  = rem - h * HD;
          int n   = row >> 10, l = row & 1023;
          out_h[(size_t)m3 * QKVSZ +
                (((size_t)(n * HEADS + h)) * LTOK + l) * HDP + dd] = (half_t)v;
        } else if (EPI == EPI_PROJ) {
          int n = row >> 10, l = row & 1023;
          float xv = res[((size_t)n * CDIM + col) * LTOK + l];   // x NCHW
          out_f[(size_t)row * CDIM + col] = xv + v;              // x2 token-major
        } else if (EPI == EPI_GELU) {
          float gv = 0.5f * v * (1.f + erff(v * 0.70710678f));
          out_h[(size_t)row * Nc + col] = (half_t)gv;
        } else {  // EPI_MLP2
          int n = row >> 10, l = row & 1023;
          float xv = res[(size_t)row * CDIM + col];              // x2 token-major
          out_f[((size_t)n * CDIM + col) * LTOK + l] = xv + v;   // d_out NCHW
        }
      }
}

// ---------------------------------------------------------------------------
// Flash attention with relative-position bias.
// Block = one (n,h) batch x 128 query rows; wave owns 16 rows; key tiles of 32.
// rel-pos tables staged in LDS (only 63 distinct dh/dw values for 32x32).
// ---------------------------------------------------------------------------
__global__ __launch_bounds__(256) void attn_kernel(const half_t* __restrict__ q,
                                                   const half_t* __restrict__ k,
                                                   const half_t* __restrict__ v,
                                                   const float* __restrict__ relh,
                                                   const float* __restrict__ relw,
                                                   half_t* __restrict__ attn_out) {
  __shared__ __align__(16) half_t Klds[32 * 64];   // [key][d]
  __shared__ __align__(16) half_t Vtlds[64 * 32];  // [d][key]
  __shared__ __align__(16) half_t Plds[8 * 16 * 32];
  __shared__ float relh_s[64];
  __shared__ float relw_s[64];

  const int tid   = threadIdx.x;
  const int lane  = tid & 31;
  const int wv    = tid >> 5;
  const int batch = blockIdx.x >> 3;      // n*HEADS + h
  const int n     = batch >> 3;
  const int h     = batch & 7;
  const int qt    = blockIdx.x & 7;
  const int qr0   = qt * 128 + wv * 16;
  const int mrow  = lane & 15;

  // stage rel-pos bias tables for this head: delta in [-31,31] -> idx d+31
  if (tid < 63) {
    relh_s[tid] = relh[(tid - 31 + 255) * HEADS + h];
    relw_s[tid] = relw[(tid - 31 + 255) * HEADS + h];
  }

  const size_t bbase = (size_t)batch * LTOK * HDP;

  // Q fragments for this wave's 16 rows (K = d, padded to 64)
  v16h qa0 = frag_ld(q + bbase + (size_t)(qr0 + mrow) * HDP, lane);
  v16h qa1 = frag_ld(q + bbase + (size_t)(qr0 + mrow) * HDP + 32, lane);

  float rmax[8], rsum[8];
  v8f o[4];
#pragma unroll
  for (int j = 0; j < 8; ++j) { rmax[j] = -1e30f; rsum[j] = 0.f; }
#pragma unroll
  for (int dt = 0; dt < 4; ++dt) o[dt] = v8f_zero();

  const float scale   = 0.14433756729740643f;  // 48^-0.5
  const int   lq_base = qr0 + ((lane >> 4) << 3);
  const int   lk_lane = lane & 15;

  for (int kt = 0; kt < 32; ++kt) {
    {  // stage K tile [32][64] (async) and transposed V tile [64][32]
      int r = tid >> 3, c = (tid & 7) * 8;
      async_cp16(Klds + r * 64 + c,
                 k + bbase + (size_t)(kt * 32 + r) * HDP + c);
      v8h vv = *(const v8h*)(v + bbase + (size_t)(kt * 32 + r) * HDP + c);
#pragma unroll
      for (int i = 0; i < 8; ++i) Vtlds[(c + i) * 32 + r] = vv[i];
    }
    async_wait();
    __syncthreads();

    v8f s[2] = {v8f_zero(), v8f_zero()};
#pragma unroll
    for (int nt = 0; nt < 2; ++nt) {
      const half_t* krow = Klds + (nt * 16 + mrow) * 64;
      v16h b0 = frag_ld(krow, lane);
      v16h b1 = frag_ld(krow + 32, lane);
      s[nt] = wmma_f16(qa0, b0, s[nt]);
      s[nt] = wmma_f16(qa1, b1, s[nt]);
    }

    // bias + online softmax (row reductions across the 16-lane half-wave)
#pragma unroll
    for (int j = 0; j < 8; ++j) {
      int   lq  = lq_base + j;
      int   chq = lq >> 5, cwq = lq & 31;
      float sv[2];
#pragma unroll
      for (int nt = 0; nt < 2; ++nt) {
        int lk  = kt * 32 + nt * 16 + lk_lane;
        int chk = lk >> 5, cwk = lk & 31;
        float bb = relh_s[chq - chk + 31] + relw_s[cwq - cwk + 31];
        sv[nt] = s[nt][j] * scale + bb;
      }
      float tmax = fmaxf(sv[0], sv[1]);
#pragma unroll
      for (int m = 1; m <= 8; m <<= 1) tmax = fmaxf(tmax, __shfl_xor(tmax, m, 32));
      float nm   = fmaxf(rmax[j], tmax);
      float corr = __expf(rmax[j] - nm);
      rmax[j] = nm;
      float p0 = __expf(sv[0] - nm);
      float p1 = __expf(sv[1] - nm);
      float ps = p0 + p1;
#pragma unroll
      for (int m = 1; m <= 8; m <<= 1) ps += __shfl_xor(ps, m, 32);
      rsum[j] = rsum[j] * corr + ps;
#pragma unroll
      for (int dt = 0; dt < 4; ++dt) o[dt][j] = o[dt][j] * corr;
      int prow = wv * 16 + ((lane >> 4) << 3) + j;
      Plds[prow * 32 + lk_lane]      = (half_t)p0;
      Plds[prow * 32 + 16 + lk_lane] = (half_t)p1;
    }

    // P @ V : A = P[16x32], B = Vt rows (d-major, key-contiguous)
    v16h pa = frag_ld(Plds + (wv * 16 + mrow) * 32, lane);
#pragma unroll
    for (int dt = 0; dt < 4; ++dt) {
      v16h bv = frag_ld(Vtlds + (dt * 16 + mrow) * 32, lane);
      o[dt] = wmma_f16(pa, bv, o[dt]);
    }
    __syncthreads();
  }

  // normalize + store [token][h*48+d] (d<48; padded cols dropped)
#pragma unroll
  for (int dt = 0; dt < 3; ++dt)
#pragma unroll
    for (int j = 0; j < 8; ++j) {
      int   d   = dt * 16 + lk_lane;
      int   lq  = lq_base + j;
      float val = o[dt][j] / rsum[j];
      attn_out[((size_t)(n * LTOK + lq)) * CDIM + h * HD + d] = (half_t)val;
    }
}

// ---------------------------------------------------------------------------
// Launch
// ---------------------------------------------------------------------------
static inline size_t alignup(size_t x) { return (x + 255) & ~(size_t)255; }

extern "C" void kernel_launch(void* const* d_in, const int* in_sizes, int n_in,
                              void* d_out, int out_size, void* d_ws, size_t ws_size,
                              hipStream_t stream) {
  (void)in_sizes; (void)n_in; (void)out_size; (void)ws_size;

  const float* x        = (const float*)d_in[0];
  const float* pos      = (const float*)d_in[1];
  const float* w_qkv    = (const float*)d_in[2];
  const float* b_qkv    = (const float*)d_in[3];
  const float* w_proj   = (const float*)d_in[4];
  const float* b_proj   = (const float*)d_in[5];
  const float* ln1_g    = (const float*)d_in[6];
  const float* ln1_b    = (const float*)d_in[7];
  const float* ln2_g    = (const float*)d_in[8];
  const float* ln2_b    = (const float*)d_in[9];
  const float* w_mlp1   = (const float*)d_in[10];
  const float* b_mlp1   = (const float*)d_in[11];
  const float* w_mlp2   = (const float*)d_in[12];
  const float* b_mlp2   = (const float*)d_in[13];
  const float* rel_h    = (const float*)d_in[14];
  const float* rel_w    = (const float*)d_in[15];
  float* out = (float*)d_out;

  char* ws = (char*)d_ws;
  size_t off = 0;
  half_t* xf      = (half_t*)(ws + off); off = alignup(off + (size_t)MTOT * CDIM * 2);
  half_t* wqkv_t  = (half_t*)(ws + off); off = alignup(off + (size_t)3 * CDIM * CDIM * 2);
  half_t* wproj_t = (half_t*)(ws + off); off = alignup(off + (size_t)CDIM * CDIM * 2);
  half_t* wmlp1_t = (half_t*)(ws + off); off = alignup(off + (size_t)CDIM * DMLP * 2);
  half_t* wmlp2_t = (half_t*)(ws + off); off = alignup(off + (size_t)CDIM * DMLP * 2);
  half_t* qkvbuf  = (half_t*)(ws + off); off = alignup(off + 3 * QKVSZ * 2);
  half_t* attn_o  = (half_t*)(ws + off); off = alignup(off + (size_t)MTOT * CDIM * 2);
  float*  x2      = (float*)(ws + off);  off = alignup(off + (size_t)MTOT * CDIM * 4);
  half_t* y2      = (half_t*)(ws + off); off = alignup(off + (size_t)MTOT * CDIM * 2);
  half_t* mlp_h   = (half_t*)(ws + off); off = alignup(off + (size_t)MTOT * DMLP * 2);

  // 0) zero padded q/k/v (pad cols d=48..63 must be zero for WMMA)
  {
    size_t n32 = (3 * QKVSZ * 2) / 4;
    fill0_kernel<<<dim3((unsigned)((n32 + 255) / 256)), 256, 0, stream>>>(
        (uint32_t*)qkvbuf, n32);
  }
  // 1) weight transpose + f16 convert
  {
    int e;
    e = CDIM * 3 * CDIM;
    tconv_kernel<<<(e + 255) / 256, 256, 0, stream>>>(w_qkv, wqkv_t, CDIM, 3 * CDIM);
    e = CDIM * CDIM;
    tconv_kernel<<<(e + 255) / 256, 256, 0, stream>>>(w_proj, wproj_t, CDIM, CDIM);
    e = CDIM * DMLP;
    tconv_kernel<<<(e + 255) / 256, 256, 0, stream>>>(w_mlp1, wmlp1_t, CDIM, DMLP);
    e = DMLP * CDIM;
    tconv_kernel<<<(e + 255) / 256, 256, 0, stream>>>(w_mlp2, wmlp2_t, DMLP, CDIM);
  }
  // 2) LN1 + pos_embed -> xf (f16)
  ln1_kernel<<<MTOT / 256, 256, 0, stream>>>(x, ln1_g, ln1_b, pos, xf);
  // 3) QKV projection -> padded q/k/v
  gemm_wmma<EPI_QKV><<<dim3(MTOT / 128, (3 * CDIM) / 128), 256, 0, stream>>>(
      xf, wqkv_t, b_qkv, MTOT, 3 * CDIM, CDIM, nullptr, qkvbuf, nullptr);
  // 4) attention (flash, rel-pos bias)
  attn_kernel<<<NB * HEADS * (LTOK / 128), 256, 0, stream>>>(
      qkvbuf, qkvbuf + QKVSZ, qkvbuf + 2 * QKVSZ, rel_h, rel_w, attn_o);
  // 5) output projection + residual -> x2 (token-major f32)
  gemm_wmma<EPI_PROJ><<<dim3(MTOT / 128, CDIM / 128), 256, 0, stream>>>(
      attn_o, wproj_t, b_proj, MTOT, CDIM, CDIM, x, nullptr, x2);
  // 6) LN2 -> y2 (f16)
  ln2_kernel<<<MTOT / 8, 256, 0, stream>>>(x2, ln2_g, ln2_b, y2);
  // 7) MLP1 + exact GELU -> mlp_h (f16)
  gemm_wmma<EPI_GELU><<<dim3(MTOT / 128, DMLP / 128), 256, 0, stream>>>(
      y2, wmlp1_t, b_mlp1, MTOT, DMLP, CDIM, nullptr, mlp_h, nullptr);
  // 8) MLP2 + residual -> d_out (NCHW f32)
  gemm_wmma<EPI_MLP2><<<dim3(MTOT / 128, CDIM / 128), 256, 0, stream>>>(
      mlp_h, wmlp2_t, b_mlp2, MTOT, CDIM, DMLP, x2, nullptr, out);
}
